// IGNet_60696477827610
// MI455X (gfx1250) — compile-verified, matrix operand order checked by reading, not run on hardware
//
#include <hip/hip_runtime.h>
#include <hip/hip_bf16.h>

#define NUM_VIEW 300
#define NUM_ANGLE 12
#define NUM_DEPTH 4
#define VAD (NUM_VIEW*NUM_ANGLE*NUM_DEPTH)   // 14400
#define BB 2
#define NS 512
#define NG 1024
#define CDIM 512
#define OSC 72000
#define PI_F 3.14159265358979323846

typedef __attribute__((ext_vector_type(16))) _Float16 v16h;
typedef __attribute__((ext_vector_type(8)))  _Float16 v8h;
typedef __attribute__((ext_vector_type(8)))  float    v8f;
typedef __attribute__((ext_vector_type(4)))  float    v4f;
typedef __attribute__((ext_vector_type(4)))  int      v4i;

// ---------------------------------------------------------------------------
// f32 -> f16 bulk conversion (weights), 8 elements/thread.
// Source read once (non-temporal); dest is L2-resident working set (regular).
// ---------------------------------------------------------------------------
__global__ __launch_bounds__(256)
void cvt_f16_kernel(const float* __restrict__ src, _Float16* __restrict__ dst, int n8) {
  int id = blockIdx.x*blockDim.x + threadIdx.x;
  if (id >= n8) return;
  size_t base = (size_t)id * 8;
  v4f a = __builtin_nontemporal_load((const v4f*)(src + base));
  v4f b = __builtin_nontemporal_load((const v4f*)(src + base + 4));
  v8h h;
  h[0] = (_Float16)a[0]; h[1] = (_Float16)a[1]; h[2] = (_Float16)a[2]; h[3] = (_Float16)a[3];
  h[4] = (_Float16)b[0]; h[5] = (_Float16)b[1]; h[6] = (_Float16)b[2]; h[7] = (_Float16)b[3];
  *(v8h*)(dst + base) = h;
}

// ---------------------------------------------------------------------------
// GEMM 1: f = relu(W1 @ S + b1); stores TRANSPOSED f16: fT[b][n][c'] (K-contig)
// orientation: M = o (rows of W1), N = n (seed columns)
// ---------------------------------------------------------------------------
__global__ __launch_bounds__(128)
void gemm1_kernel(const float* __restrict__ W1, const float* __restrict__ S,
                  const float* __restrict__ b1, _Float16* __restrict__ FT) {
  __shared__ __align__(16) _Float16 lds_a[64*32];   // [m=o][k]
  __shared__ __align__(16) _Float16 lds_b[16*32];   // [n][k] (transposed)
  const int tid  = threadIdx.x;
  const int wave = tid >> 5, lane = tid & 31;
  const int m0 = blockIdx.x * 64, n0 = blockIdx.y * 16, b = blockIdx.z;
  const float* Sb = S + (size_t)b * CDIM * NS;
  v8f acc = {};
  const int bk   = (lane >> 4) * 8;
  const int mloc = 16*wave + (lane & 15);
  const int nl   = lane & 15;
  for (int k0 = 0; k0 < CDIM; k0 += 32) {
    #pragma unroll
    for (int q = 0; q < 4; ++q) {
      int idx = tid + 128*q;
      int r = idx >> 3, cc = (idx & 7) * 4;
      v4f v = *(const v4f*)(W1 + (size_t)(m0 + r)*CDIM + k0 + cc);
      lds_a[r*32+cc+0] = (_Float16)v[0]; lds_a[r*32+cc+1] = (_Float16)v[1];
      lds_a[r*32+cc+2] = (_Float16)v[2]; lds_a[r*32+cc+3] = (_Float16)v[3];
    }
    {
      int kk = tid >> 2, ng = (tid & 3) * 4;
      v4f v = *(const v4f*)(Sb + (size_t)(k0 + kk)*NS + n0 + ng);
      lds_b[(ng+0)*32+kk] = (_Float16)v[0]; lds_b[(ng+1)*32+kk] = (_Float16)v[1];
      lds_b[(ng+2)*32+kk] = (_Float16)v[2]; lds_b[(ng+3)*32+kk] = (_Float16)v[3];
    }
    __syncthreads();
    v8h alo = *(const v8h*)(lds_a + mloc*32 + bk);
    v8h ahi = *(const v8h*)(lds_a + mloc*32 + bk + 16);
    v16h afr = __builtin_shufflevector(alo, ahi, 0,1,2,3,4,5,6,7,8,9,10,11,12,13,14,15);
    v8h blo = *(const v8h*)(lds_b + nl*32 + bk);
    v8h bhi = *(const v8h*)(lds_b + nl*32 + bk + 16);
    v16h bfr = __builtin_shufflevector(blo, bhi, 0,1,2,3,4,5,6,7,8,9,10,11,12,13,14,15);
    acc = __builtin_amdgcn_wmma_f32_16x16x32_f16(false, afr, false, bfr,
                                                 (short)0, acc, false, false);
    __syncthreads();
  }
  // epilogue: relu(+bias), store transposed fT[n][o] (8 contiguous o per lane)
  const int n = n0 + nl;
  const int obase = m0 + 16*wave + 8*(lane >> 4);
  v8h hv;
  #pragma unroll
  for (int r = 0; r < 8; ++r) {
    float v = acc[r] + b1[obase + r];
    v = v > 0.0f ? v : 0.0f;
    hv[r] = (_Float16)v;
  }
  *(v8h*)(FT + (size_t)b*NS*CDIM + (size_t)n*CDIM + obase) = hv;
}

// ---------------------------------------------------------------------------
// GEMM 2/3: out[b][n][o] = sum_c wf16[o][c] * fT[b][n][c] + bias[o]
// Double-buffered GLOBAL_LOAD_ASYNC_TO_LDS_B128 staging (ASYNCcnt) overlapped
// with WMMA; block tile 64x32, 2 WMMAs/wave/K-step; non-temporal epilogue.
// ---------------------------------------------------------------------------
__device__ __forceinline__ void stage_tile_async(const _Float16* __restrict__ Fb,
                                                 const _Float16* __restrict__ WH,
                                                 int m0, int o0, int k0, int tid,
                                                 unsigned a_off, unsigned b_off) {
  #pragma unroll
  for (int q = 0; q < 2; ++q) {
    int idx = tid + 128*q;
    int r = idx >> 2, cc = (idx & 3) * 8;
    const _Float16* gp = Fb + (size_t)(m0 + r)*CDIM + k0 + cc;
    unsigned loff = a_off + (unsigned)(r*64 + cc*2);
    asm volatile("global_load_async_to_lds_b128 %0, %1, off"
                 :: "v"(loff), "v"(gp) : "memory");
  }
  {
    int r = tid >> 2, cc = (tid & 3) * 8;
    const _Float16* gp = WH + (size_t)(o0 + r)*CDIM + k0 + cc;
    unsigned loff = b_off + (unsigned)(r*64 + cc*2);
    asm volatile("global_load_async_to_lds_b128 %0, %1, off"
                 :: "v"(loff), "v"(gp) : "memory");
  }
}

__global__ __launch_bounds__(128)
void gemm2_kernel(const _Float16* __restrict__ FT, const _Float16* __restrict__ WH,
                  const float* __restrict__ bias, float* __restrict__ out, int O) {
  __shared__ __align__(16) _Float16 lds_a[2][64*32];   // [m=n][k=c]
  __shared__ __align__(16) _Float16 lds_b[2][32*32];   // [o][k=c]
  const int tid  = threadIdx.x;
  const int wave = tid >> 5, lane = tid & 31;
  const int m0 = blockIdx.x * 64, o0 = blockIdx.y * 32, b = blockIdx.z;
  const _Float16* Fb = FT + (size_t)b * NS * CDIM;
  v8f acc0 = {}, acc1 = {};
  const int bk   = (lane >> 4) * 8;
  const int mloc = 16*wave + (lane & 15);
  const int nl   = lane & 15;
  unsigned a_off[2], b_off[2];
  a_off[0] = (unsigned)(size_t)(&lds_a[0][0]); a_off[1] = (unsigned)(size_t)(&lds_a[1][0]);
  b_off[0] = (unsigned)(size_t)(&lds_b[0][0]); b_off[1] = (unsigned)(size_t)(&lds_b[1][0]);

  // prologue: fill buffer 0 with K-tile 0
  stage_tile_async(Fb, WH, m0, o0, 0, tid, a_off[0], b_off[0]);
  asm volatile("s_wait_asynccnt 0" ::: "memory");
  __syncthreads();

  for (int k0 = 0; k0 < CDIM; k0 += 32) {
    const int cur = (k0 >> 5) & 1;
    // prefetch next K-tile into the other buffer (overlaps WMMA below)
    if (k0 + 32 < CDIM)
      stage_tile_async(Fb, WH, m0, o0, k0 + 32, tid, a_off[cur^1], b_off[cur^1]);
    const _Float16* la = lds_a[cur];
    const _Float16* lb = lds_b[cur];
    v8h alo = *(const v8h*)(la + mloc*32 + bk);
    v8h ahi = *(const v8h*)(la + mloc*32 + bk + 16);
    v16h afr = __builtin_shufflevector(alo, ahi, 0,1,2,3,4,5,6,7,8,9,10,11,12,13,14,15);
    v8h b0l = *(const v8h*)(lb + nl*32 + bk);
    v8h b0h = *(const v8h*)(lb + nl*32 + bk + 16);
    v16h bf0 = __builtin_shufflevector(b0l, b0h, 0,1,2,3,4,5,6,7,8,9,10,11,12,13,14,15);
    v8h b1l = *(const v8h*)(lb + (16+nl)*32 + bk);
    v8h b1h = *(const v8h*)(lb + (16+nl)*32 + bk + 16);
    v16h bf1 = __builtin_shufflevector(b1l, b1h, 0,1,2,3,4,5,6,7,8,9,10,11,12,13,14,15);
    acc0 = __builtin_amdgcn_wmma_f32_16x16x32_f16(false, afr, false, bf0,
                                                  (short)0, acc0, false, false);
    acc1 = __builtin_amdgcn_wmma_f32_16x16x32_f16(false, afr, false, bf1,
                                                  (short)0, acc1, false, false);
    asm volatile("s_wait_asynccnt 0" ::: "memory");
    __syncthreads();
  }
  const int oA = o0 + nl, oB = o0 + 16 + nl;
  const float bA = bias[oA], bW = bias[oB];
  #pragma unroll
  for (int r = 0; r < 8; ++r) {
    int m = m0 + 16*wave + r + 8*(lane >> 4);
    size_t row = (size_t)b*NS*(size_t)O + (size_t)m*O;
    __builtin_nontemporal_store(acc0[r] + bA, &out[row + oA]);
    __builtin_nontemporal_store(acc1[r] + bW, &out[row + oB]);
  }
}

// ---------------------------------------------------------------------------
// Geometry
// ---------------------------------------------------------------------------
__device__ inline void rot_kp(const float* R, float lx, float ly, float* o) {
  o[0] = R[0]*lx + R[1]*ly;
  o[1] = R[3]*lx + R[4]*ly;
  o[2] = R[6]*lx + R[7]*ly;
}

__device__ inline void view_rot(const float* views, int v, float ang, float* R) {
  float tx = -views[v*3+0], ty = -views[v*3+1], tz = -views[v*3+2];
  float ayx = -ty, ayy = tx;
  float ny = sqrtf(ayx*ayx + ayy*ayy);
  if (ny < 1e-12f) { ayx = 0.0f; ayy = 1.0f; ny = 1.0f; }
  float nx = sqrtf(tx*tx + ty*ty + tz*tz);
  float x0 = tx/nx, x1 = ty/nx, x2 = tz/nx;
  float y0 = ayx/ny, y1 = ayy/ny, y2 = 0.0f;
  float z0 = x1*y2 - x2*y1, z1 = x2*y0 - x0*y2, z2 = x0*y1 - x1*y0;
  float c = cosf(ang), s = sinf(ang);
  R[0] = x0; R[1] = c*y0 + s*z0; R[2] = -s*y0 + c*z0;
  R[3] = x1; R[4] = c*y1 + s*z1; R[5] = -s*y1 + c*z1;
  R[6] = x2; R[7] = c*y2 + s*z2; R[8] = -s*y2 + c*z2;
}

__global__ void geo_views_rot(float* __restrict__ views, float* __restrict__ grot,
                              float* __restrict__ pk) {
  int tid = threadIdx.x;
  for (int v = tid; v < NUM_VIEW; v += blockDim.x) {
    double phi = (sqrt(5.0) - 1.0) / 2.0;
    double zi = (2.0*v + 1.0) / NUM_VIEW - 1.0;
    double r2 = 1.0 - zi*zi; if (r2 < 0.0) r2 = 0.0;
    double r = sqrt(r2);
    double ang = 2.0 * PI_F * (double)v * phi;
    views[v*3+0] = (float)(r * cos(ang));
    views[v*3+1] = (float)(r * sin(ang));
    views[v*3+2] = (float)zi;
  }
  __threadfence_block();
  __syncthreads();
  for (int idx = tid; idx < NUM_VIEW*NUM_ANGLE; idx += blockDim.x) {
    int v = idx / NUM_ANGLE, a = idx % NUM_ANGLE;
    float R[9];
    view_rot(views, v, (float)(PI_F / NUM_ANGLE) * (float)a, R);
    float* G = grot + idx*9;
    #pragma unroll
    for (int i = 0; i < 9; ++i) G[i] = R[i];
    float* P = pk + idx*12;
    P[0] = 0.0f; P[1] = 0.0f; P[2] = 0.0f;
    rot_kp(R,  0.02f,  0.01f, P+3);
    rot_kp(R,  0.02f, -0.01f, P+6);
    rot_kp(R, -0.02f,  0.00f, P+9);
  }
}

__global__ __launch_bounds__(384)
void geo_batch(const float* __restrict__ pose, const float* __restrict__ gpts,
               const float* __restrict__ views, const float* __restrict__ grot,
               const float* __restrict__ pk,
               int* __restrict__ vinds, int* __restrict__ gidx,
               float* __restrict__ tkbuf, float* __restrict__ tksbuf,
               float* __restrict__ gpt_trans) {
  const int b = blockIdx.x, tid = threadIdx.x;
  __shared__ float Rm[9], tv[3];
  __shared__ float vt[NUM_VIEW*3];
  if (tid < 12) {
    int i = tid / 4, j = tid % 4;
    float val = pose[b*12 + i*4 + j];
    if (j < 3) Rm[i*3+j] = val; else tv[i] = val;
  }
  __syncthreads();
  for (int v = tid; v < NUM_VIEW; v += blockDim.x) {
    float px = views[v*3], py = views[v*3+1], pz = views[v*3+2];
    vt[v*3+0] = Rm[0]*px + Rm[1]*py + Rm[2]*pz;
    vt[v*3+1] = Rm[3]*px + Rm[4]*py + Rm[5]*pz;
    vt[v*3+2] = Rm[6]*px + Rm[7]*py + Rm[8]*pz;
  }
  for (int g = tid; g < NG; g += blockDim.x) {
    float px = gpts[((size_t)b*NG + g)*3+0];
    float py = gpts[((size_t)b*NG + g)*3+1];
    float pz = gpts[((size_t)b*NG + g)*3+2];
    gpt_trans[((size_t)b*NG + g)*3+0] = Rm[0]*px + Rm[1]*py + Rm[2]*pz + tv[0];
    gpt_trans[((size_t)b*NG + g)*3+1] = Rm[3]*px + Rm[4]*py + Rm[5]*pz + tv[1];
    gpt_trans[((size_t)b*NG + g)*3+2] = Rm[6]*px + Rm[7]*py + Rm[8]*pz + tv[2];
  }
  __syncthreads();
  for (int v = tid; v < NUM_VIEW; v += blockDim.x) {
    float px = views[v*3], py = views[v*3+1], pz = views[v*3+2];
    float bd = 1e30f; int best = 0;
    for (int j = 0; j < NUM_VIEW; ++j) {
      float dx = px - vt[j*3], dy = py - vt[j*3+1], dz = pz - vt[j*3+2];
      float d = dx*dx + dy*dy + dz*dz;
      if (d < bd) { bd = d; best = j; }
    }
    vinds[b*NUM_VIEW + v] = best;
  }
  __threadfence_block();
  __syncthreads();
  for (int idx = tid; idx < NUM_VIEW*NUM_ANGLE; idx += blockDim.x) {
    int v = idx / NUM_ANGLE, a = idx % NUM_ANGLE;
    int vi = vinds[b*NUM_VIEW + v];
    const float* G = grot + (vi*NUM_ANGLE + a)*9;
    float Rt[9];
    #pragma unroll
    for (int i = 0; i < 3; ++i)
      #pragma unroll
      for (int j = 0; j < 3; ++j)
        Rt[i*3+j] = Rm[i*3+0]*G[0*3+j] + Rm[i*3+1]*G[1*3+j] + Rm[i*3+2]*G[2*3+j];
    float kp1[3], kp2[3], kp3[3];
    rot_kp(Rt,  0.02f,  0.01f, kp1);
    rot_kp(Rt,  0.02f, -0.01f, kp2);
    rot_kp(Rt, -0.02f,  0.00f, kp3);
    float* T  = tkbuf  + ((size_t)b*NUM_VIEW*NUM_ANGLE + idx)*12;
    float* Ts = tksbuf + ((size_t)b*NUM_VIEW*NUM_ANGLE + idx)*12;
    T[0]=0; T[1]=0; T[2]=0;           Ts[0]=0; Ts[1]=0; Ts[2]=0;
    T[3]=kp1[0]; T[4]=kp1[1]; T[5]=kp1[2];   Ts[3]=kp2[0]; Ts[4]=kp2[1]; Ts[5]=kp2[2];
    T[6]=kp2[0]; T[7]=kp2[1]; T[8]=kp2[2];   Ts[6]=kp1[0]; Ts[7]=kp1[1]; Ts[8]=kp1[2];
    T[9]=kp3[0]; T[10]=kp3[1]; T[11]=kp3[2]; Ts[9]=kp3[0]; Ts[10]=kp3[1]; Ts[11]=kp3[2];
  }
  __threadfence_block();
  __syncthreads();
  for (int idx = tid; idx < NUM_VIEW*NUM_ANGLE; idx += blockDim.x) {
    int v = idx / NUM_ANGLE;
    const float* p1 = pk + idx*12;
    float mind = 1e30f, minds = 1e30f; int ad = 0, ads = 0;
    for (int a2 = 0; a2 < NUM_ANGLE; ++a2) {
      const float* T  = tkbuf  + ((size_t)b*NUM_VIEW*NUM_ANGLE + v*NUM_ANGLE + a2)*12;
      const float* Ts = tksbuf + ((size_t)b*NUM_VIEW*NUM_ANGLE + v*NUM_ANGLE + a2)*12;
      float d = 0.0f, ds = 0.0f;
      #pragma unroll
      for (int k = 0; k < 12; ++k) {
        float e = p1[k] - T[k];  d  += e*e;
        float es = p1[k] - Ts[k]; ds += es*es;
      }
      if (d  < mind)  { mind  = d;  ad  = a2; }
      if (ds < minds) { minds = ds; ads = a2; }
    }
    int va = (mind < minds) ? ad : ads;
    gidx[b*NUM_VIEW*NUM_ANGLE + idx] =
        vinds[b*NUM_VIEW + v]*(NUM_ANGLE*NUM_DEPTH) + va*NUM_DEPTH;
  }
}

__global__ void geo_nn(const float* __restrict__ pc, const float* __restrict__ gpt_trans,
                       int* __restrict__ nn, float* __restrict__ pts_out) {
  int id = blockIdx.x*blockDim.x + threadIdx.x;
  if (id >= BB*NS) return;
  int b = id / NS;
  float sx = pc[(size_t)id*3], sy = pc[(size_t)id*3+1], sz = pc[(size_t)id*3+2];
  int best = 0; float bd = 1e30f;
  const float* base = gpt_trans + (size_t)b*NG*3;
  for (int g = 0; g < NG; ++g) {
    float dx = sx - base[g*3], dy = sy - base[g*3+1], dz = sz - base[g*3+2];
    float d = dx*dx + dy*dy + dz*dz;
    if (d < bd) { bd = d; best = g; }
  }
  nn[id] = best;
  pts_out[(size_t)id*3+0] = base[best*3+0];
  pts_out[(size_t)id*3+1] = base[best*3+1];
  pts_out[(size_t)id*3+2] = base[best*3+2];
}

__global__ void geo_gather(const float* __restrict__ labels, const float* __restrict__ offsets,
                           const int* __restrict__ nn, const int* __restrict__ gidx,
                           float* __restrict__ out_scores, float* __restrict__ out_widths,
                           int* __restrict__ out_wid, int* __restrict__ out_sid) {
  int id = blockIdx.x*blockDim.x + threadIdx.x;
  const int total = BB*NS*NUM_VIEW*NUM_ANGLE;
  if (id >= total) return;
  int va = id % (NUM_VIEW*NUM_ANGLE);
  int rest = id / (NUM_VIEW*NUM_ANGLE);
  int s = rest % NS, b = rest / NS;
  int n = nn[b*NS + s];
  int g = gidx[b*NUM_VIEW*NUM_ANGLE + va];
  size_t base = ((size_t)(b*NG + n))*VAD + (size_t)g;
  v4f lab = __builtin_nontemporal_load((const v4f*)(labels + base));
  v4f off = __builtin_nontemporal_load((const v4f*)(offsets + base));
  v4f so; v4i wid, sid;
  #pragma unroll
  for (int d = 0; d < 4; ++d) {
    float sc = lab[d], wd = off[d];
    float m = (sc > 0.0f && wd <= 0.1f) ? (1.1f - sc) : 0.0f;
    so[d] = m;
    wid[d] = (wd > 0.02f) + (wd > 0.04f) + (wd > 0.06f) + (wd > 0.08f);
    sid[d] = (m > 0.2f) + (m > 0.4f) + (m > 0.6f) + (m > 0.8f);
  }
  size_t obase = ((size_t)(b*NS + s))*VAD + (size_t)va*4;
  __builtin_nontemporal_store(so,  (v4f*)(out_scores + obase));
  __builtin_nontemporal_store(off, (v4f*)(out_widths + obase));
  __builtin_nontemporal_store(wid, (v4i*)(out_wid + obase));
  __builtin_nontemporal_store(sid, (v4i*)(out_sid + obase));
}

// ---------------------------------------------------------------------------
extern "C" void kernel_launch(void* const* d_in, const int* in_sizes, int n_in,
                              void* d_out, int out_size, void* d_ws, size_t ws_size,
                              hipStream_t stream) {
  (void)in_sizes; (void)n_in; (void)out_size; (void)ws_size;
  const float* seed    = (const float*)d_in[0];
  const float* pc      = (const float*)d_in[1];
  const float* pose    = (const float*)d_in[2];
  const float* gpts    = (const float*)d_in[3];
  const float* labels  = (const float*)d_in[4];
  const float* offsets = (const float*)d_in[5];
  const float* w1      = (const float*)d_in[6];
  const float* b1      = (const float*)d_in[7];
  const float* w_score = (const float*)d_in[8];
  const float* b_score = (const float*)d_in[9];
  const float* w_width = (const float*)d_in[10];
  const float* b_width = (const float*)d_in[11];

  char* ws = (char*)d_ws;
  size_t off = 0;
  auto take = [&](size_t bytes) { char* p = ws + off; off = (off + bytes + 15) & ~(size_t)15; return p; };
  _Float16* fT      = (_Float16*)take((size_t)BB*NS*CDIM*2);           // 1 MB, [b][n][c]
  _Float16* whs     = (_Float16*)take((size_t)OSC*CDIM*2);             // 73.7 MB
  _Float16* whw     = (_Float16*)take((size_t)VAD*CDIM*2);             // 14.7 MB
  float* ws_views   = (float*)take(NUM_VIEW*3*4);
  float* ws_grot    = (float*)take(NUM_VIEW*NUM_ANGLE*9*4);
  float* ws_pk      = (float*)take(NUM_VIEW*NUM_ANGLE*12*4);
  int*   ws_vinds   = (int*)  take(BB*NUM_VIEW*4);
  int*   ws_gidx    = (int*)  take(BB*NUM_VIEW*NUM_ANGLE*4);
  float* ws_tk      = (float*)take((size_t)BB*NUM_VIEW*NUM_ANGLE*12*4);
  float* ws_tks     = (float*)take((size_t)BB*NUM_VIEW*NUM_ANGLE*12*4);
  float* ws_gpt     = (float*)take((size_t)BB*NG*3*4);
  int*   ws_nn      = (int*)  take(BB*NS*4);

  float* out = (float*)d_out;
  const size_t O1 = 0;                         // score_pred: 2*512*72000
  const size_t O2 = 73728000;                  // width_pred: 2*512*14400
  const size_t O3 = 88473600;                  // pts: 2*512*3
  const size_t O4 = 88476672;                  // scores
  const size_t O5 = 103222272;                 // widths
  const size_t O6 = 117967872;                 // width_ids (int32)
  const size_t O7 = 132713472;                 // score_ids (int32)

  // geometry chain
  geo_views_rot<<<1, 512, 0, stream>>>(ws_views, ws_grot, ws_pk);
  geo_batch<<<BB, 384, 0, stream>>>(pose, gpts, ws_views, ws_grot, ws_pk,
                                    ws_vinds, ws_gidx, ws_tk, ws_tks, ws_gpt);
  geo_nn<<<(BB*NS + 255)/256, 256, 0, stream>>>(pc, ws_gpt, ws_nn, out + O3);
  {
    int total = BB*NS*NUM_VIEW*NUM_ANGLE;
    geo_gather<<<(total + 255)/256, 256, 0, stream>>>(labels, offsets, ws_nn, ws_gidx,
                                                      out + O4, out + O5,
                                                      (int*)(out + O6), (int*)(out + O7));
  }
  // weight conversion (once per call) then WMMA GEMM chain
  {
    int n8s = OSC*CDIM/8, n8w = VAD*CDIM/8;
    cvt_f16_kernel<<<(n8s + 255)/256, 256, 0, stream>>>(w_score, whs, n8s);
    cvt_f16_kernel<<<(n8w + 255)/256, 256, 0, stream>>>(w_width, whw, n8w);
  }
  gemm1_kernel<<<dim3(CDIM/64, NS/16, BB), 128, 0, stream>>>(w1, seed, b1, fT);
  gemm2_kernel<<<dim3(NS/64, OSC/32, BB), 128, 0, stream>>>(fT, whs, b_score, out + O1, OSC);
  gemm2_kernel<<<dim3(NS/64, VAD/32, BB), 128, 0, stream>>>(fT, whw, b_width, out + O2, VAD);
}
